// Attention_67628555043143
// MI455X (gfx1250) — compile-verified, matrix operand order checked by reading, not run on hardware
//
#include <hip/hip_runtime.h>

// ---------------------------------------------------------------------------
// Problem constants: B=2, T=2048, DIM=2048, H=32, KV=8, D=64
// ---------------------------------------------------------------------------
#define BATCH   2
#define SEQ     2048
#define DIM     2048
#define NH      32
#define NKV     8
#define HD      64

typedef __attribute__((ext_vector_type(16))) __bf16 v16bf;
typedef __attribute__((ext_vector_type(8)))  __bf16 v8bf;
typedef __attribute__((ext_vector_type(4)))  __bf16 v4bf;
typedef __attribute__((ext_vector_type(8)))  float  v8f;
typedef __attribute__((ext_vector_type(4)))  float  v4f;

typedef __attribute__((address_space(3))) __bf16 lbf16;

// GCC-style int4 vector: matches the async-copy builtin's parameter type.
typedef int v4i __attribute__((vector_size(16)));
typedef __attribute__((address_space(1))) v4i gv4i;
typedef __attribute__((address_space(3))) v4i lv4i;

#if __has_builtin(__builtin_amdgcn_global_load_async_to_lds_b128)
#define HAS_ASYNC 1
#else
#define HAS_ASYNC 0
#endif

// 16-byte global -> LDS copy.  Async on CDNA5 (ASYNCcnt-tracked), sync fallback.
__device__ __forceinline__ void cp16(lbf16* ldst, const __bf16* gsrc) {
#if HAS_ASYNC
    __builtin_amdgcn_global_load_async_to_lds_b128(
        (gv4i*)gsrc, (lv4i*)ldst, 0, 0);
#else
    typedef __attribute__((address_space(3))) v8bf lv8bf;
    *(lv8bf*)ldst = *(const v8bf*)gsrc;
#endif
}

__device__ __forceinline__ void wait_async0() {
#if HAS_ASYNC
#if __has_builtin(__builtin_amdgcn_s_wait_asynccnt)
    __builtin_amdgcn_s_wait_asynccnt(0);
#else
    asm volatile("s_wait_asynccnt 0" ::: "memory");
#endif
#endif
}

__device__ __forceinline__ v8f zero8() {
    v8f z;
    for (int i = 0; i < 8; ++i) z[i] = 0.0f;
    return z;
}

__device__ __forceinline__ v16bf frag_cat(v8bf lo, v8bf hi) {
    return __builtin_shufflevector(lo, hi, 0,1,2,3,4,5,6,7,8,9,10,11,12,13,14,15);
}

// A/B fragment from a row-major bf16 LDS tile (ISA 16-bit layout: lanes 0-15
// take K 0-7 & 16-23, lanes 16-31 take K 8-15 & 24-31 -> two 16B chunks).
__device__ __forceinline__ v16bf lds_frag(const __bf16* row_ptr, int lane) {
    const int kb = (lane & 16) ? 8 : 0;
    v8bf lo = *(const v8bf*)(row_ptr + kb);
    v8bf hi = *(const v8bf*)(row_ptr + kb + 16);
    return frag_cat(lo, hi);
}

// ---------------------------------------------------------------------------
// Pre-pass: fp32 -> bf16 elementwise (x)
// ---------------------------------------------------------------------------
__global__ __launch_bounds__(256) void cvt_bf16(
    const float* __restrict__ in, __bf16* __restrict__ out, int total4)
{
    int c = blockIdx.x * 256 + threadIdx.x;
    if (c >= total4) return;
    v4f f = *(const v4f*)(in + (size_t)c * 4);
    *(v4bf*)(out + (size_t)c * 4) = __builtin_convertvector(f, v4bf);
}

// ---------------------------------------------------------------------------
// Pre-pass: W (K,N) fp32 -> Wt (N,K) bf16, LDS-tiled 32x32 transpose
// ---------------------------------------------------------------------------
__global__ __launch_bounds__(256) void transpose_cvt_w(
    const float* __restrict__ W, __bf16* __restrict__ Wt, int K, int N)
{
    __shared__ float t[32][33];
    const int tx = threadIdx.x;       // 0..31
    const int ty = threadIdx.y;       // 0..7
    const int n0 = blockIdx.x * 32;
    const int k0 = blockIdx.y * 32;
    for (int j = 0; j < 4; ++j) {
        int kk = ty * 4 + j;
        t[kk][tx] = W[(size_t)(k0 + kk) * N + n0 + tx];
    }
    __syncthreads();
    for (int j = 0; j < 4; ++j) {
        int nn = ty * 4 + j;
        Wt[(size_t)(n0 + nn) * K + k0 + tx] = (__bf16)t[tx][nn];
    }
}

// ---------------------------------------------------------------------------
// GEMM: C[M,N] = A[M,K] (bf16 row-major) x Wt[N,K]^T (bf16, k-contiguous).
// 128x128 block tile, 8 waves, wave tile 32x64, K step 32.
// Double-buffered LDS, async global->LDS staging.
// ---------------------------------------------------------------------------
template<typename OutT>
__global__ __launch_bounds__(256) void gemm_bf16_t(
    const __bf16* __restrict__ A, const __bf16* __restrict__ Wt,
    OutT* __restrict__ C, int N, int K)
{
    __shared__ __attribute__((aligned(16))) __bf16 As[2][128 * 40];
    __shared__ __attribute__((aligned(16))) __bf16 Ws[2][128 * 40];

    const int tid  = threadIdx.x;
    const int lane = tid & 31;
    const int l15  = lane & 15;
    const int half = (lane >> 4) & 1;
    const int wv   = tid >> 5;
    const int wm   = wv >> 1;
    const int wn   = wv & 1;
    const int mb   = blockIdx.y;
    const int nb   = blockIdx.x;

    const __bf16* Ag = A  + (size_t)mb * 128 * K;
    const __bf16* Wg = Wt + (size_t)nb * 128 * K;

    v8f acc[2][4];
    for (int mi = 0; mi < 2; ++mi)
        for (int ni = 0; ni < 4; ++ni)
            acc[mi][ni] = zero8();

    const int nIter = K / 32;

    // stage one tile: 128 rows x 32 bf16 = 512 x 16B chunks, 2 per thread/array
    auto issue = [&](int it, int buf) {
        const int kk = it * 32;
        lbf16* a3 = (lbf16*)&As[buf][0];
        lbf16* w3 = (lbf16*)&Ws[buf][0];
        for (int i = 0; i < 2; ++i) {
            int c = tid + i * 256;         // 0..511
            int r = c >> 2;                // row 0..127
            int j = c & 3;                 // chunk 0..3 (8 bf16 each)
            cp16(a3 + r * 40 + j * 8, Ag + (size_t)r * K + kk + j * 8);
            cp16(w3 + r * 40 + j * 8, Wg + (size_t)r * K + kk + j * 8);
        }
    };

    issue(0, 0);
    for (int it = 0; it < nIter; ++it) {
        const int buf = it & 1;
        wait_async0();
        __syncthreads();                       // tile `it` fully in LDS
        if (it + 1 < nIter) issue(it + 1, buf ^ 1);

        v16bf af[2], bf_[4];
        for (int mi = 0; mi < 2; ++mi)
            af[mi] = lds_frag(&As[buf][(wm * 32 + mi * 16 + l15) * 40], lane);
        for (int ni = 0; ni < 4; ++ni)
            bf_[ni] = lds_frag(&Ws[buf][(wn * 64 + ni * 16 + l15) * 40], lane);
        for (int mi = 0; mi < 2; ++mi)
            for (int ni = 0; ni < 4; ++ni)
                acc[mi][ni] = __builtin_amdgcn_wmma_f32_16x16x32_bf16(
                    false, af[mi], false, bf_[ni], (short)0, acc[mi][ni], false, false);
    }

    for (int mi = 0; mi < 2; ++mi)
        for (int ni = 0; ni < 4; ++ni)
            for (int r = 0; r < 8; ++r) {
                int row = mb * 128 + wm * 32 + mi * 16 + half * 8 + r;
                int col = nb * 128 + wn * 64 + ni * 16 + l15;
                C[(size_t)row * N + col] = (OutT)acc[mi][ni][r];
            }
}

// ---------------------------------------------------------------------------
// RoPE q: bf16 (B,T,NH,HD) -> bf16 (B,NH,T,HD), softmax scale 0.125 folded in
// ---------------------------------------------------------------------------
__global__ __launch_bounds__(256) void rope_q(
    const __bf16* __restrict__ in, __bf16* __restrict__ out,
    const float* __restrict__ cs, const float* __restrict__ sn, int total)
{
    int idx = blockIdx.x * 256 + threadIdx.x;
    if (idx >= total) return;
    int d2 = idx & 31;
    int r  = idx >> 5;
    int h  = r % NH;  r /= NH;
    int t  = r % SEQ;
    int b  = r / SEQ;
    const __bf16* ip = in + (((size_t)(b * SEQ + t) * NH + h) << 6) + 2 * d2;
    float t1 = (float)ip[0], t2 = (float)ip[1];
    float c = cs[t * 32 + d2], s = sn[t * 32 + d2];
    __bf16* op = out + (((size_t)(b * NH + h) * SEQ + t) << 6) + 2 * d2;
    op[0] = (__bf16)((t1 * c - t2 * s) * 0.125f);
    op[1] = (__bf16)((t1 * s + t2 * c) * 0.125f);
}

// ---------------------------------------------------------------------------
// RoPE k: fp32 (B,T,NKV,HD) -> fp32 new_k (B,NKV,T,HD) + bf16 copy same layout
// ---------------------------------------------------------------------------
__global__ __launch_bounds__(256) void rope_k(
    const float* __restrict__ in, float* __restrict__ outf,
    __bf16* __restrict__ outb,
    const float* __restrict__ cs, const float* __restrict__ sn, int total)
{
    int idx = blockIdx.x * 256 + threadIdx.x;
    if (idx >= total) return;
    int d2 = idx & 31;
    int r  = idx >> 5;
    int h  = r % NKV;  r /= NKV;
    int t  = r % SEQ;
    int b  = r / SEQ;
    const float* ip = in + (((size_t)(b * SEQ + t) * NKV + h) << 6) + 2 * d2;
    float t1 = ip[0], t2 = ip[1];
    float c = cs[t * 32 + d2], s = sn[t * 32 + d2];
    float o1 = t1 * c - t2 * s;
    float o2 = t1 * s + t2 * c;
    size_t oo = (((size_t)(b * NKV + h) * SEQ + t) << 6) + 2 * d2;
    outf[oo]     = o1;
    outf[oo + 1] = o2;
    outb[oo]     = (__bf16)o1;
    outb[oo + 1] = (__bf16)o2;
}

// ---------------------------------------------------------------------------
// v: fp32 (B,T,NKV,HD) -> fp32 new_v (B,NKV,T,HD) + bf16 TRANSPOSED (B,NKV,HD,T)
// ---------------------------------------------------------------------------
__global__ __launch_bounds__(256) void copy_v(
    const float* __restrict__ in, float* __restrict__ outf,
    __bf16* __restrict__ outb, int total)
{
    int idx = blockIdx.x * 256 + threadIdx.x;
    if (idx >= total) return;
    int d = idx & 63;
    int r = idx >> 6;
    int h = r % NKV;  r /= NKV;
    int t = r % SEQ;
    int b = r / SEQ;
    float f = in[(((size_t)(b * SEQ + t) * NKV + h) << 6) + d];
    outf[(((size_t)(b * NKV + h) * SEQ + t) << 6) + d] = f;
    outb[((size_t)(b * NKV + h) * HD + d) * SEQ + t]   = (__bf16)f;
}

// ---------------------------------------------------------------------------
// Flash attention, causal, GQA.  Grid (T/128, B*NH), 8 waves x 16 q rows.
// qTb: bf16 (B,NH,T,HD) (pre-scaled)   Kb: bf16 (B,NKV,T,HD)
// Vbt: bf16 (B,NKV,HD,T)               Ob: bf16 (B,T,NH,HD)
// K/V tiles staged via async global->LDS copies.  Causal mask applied only on
// diagonal-crossing tiles (wave-uniform branch) to keep interior tiles VALU-lean.
// ---------------------------------------------------------------------------
__global__ __launch_bounds__(256) void attn_fwd(
    const __bf16* __restrict__ qTb, const __bf16* __restrict__ Kb,
    const __bf16* __restrict__ Vbt, __bf16* __restrict__ Ob)
{
    __shared__ __attribute__((aligned(16))) __bf16 Kl[64 * 72];     // [key][d]
    __shared__ __attribute__((aligned(16))) __bf16 Vt[64 * 72];     // [d][key]
    __shared__ __attribute__((aligned(16))) __bf16 Pl[8][16 * 72];  // per-wave P

    const int tid  = threadIdx.x;
    const int lane = tid & 31;
    const int l15  = lane & 15;
    const int half = (lane >> 4) & 1;
    const int w    = tid >> 5;

    const int bh  = blockIdx.y;
    const int b   = bh >> 5;
    const int h   = bh & 31;
    const int kvh = h >> 2;
    const int qb  = blockIdx.x;
    const int qrow0 = qb * 128 + w * 16;

    // q fragments (scale already folded in by rope_q)
    v16bf aq[2];
    {
        const __bf16* qp = qTb + ((size_t)(b * NH + h) * SEQ + qrow0 + l15) * HD;
        const int kb = half ? 8 : 0;
        for (int dc = 0; dc < 2; ++dc) {
            v8bf lo = *(const v8bf*)(qp + dc * 32 + kb);
            v8bf hi = *(const v8bf*)(qp + dc * 32 + kb + 16);
            aq[dc] = frag_cat(lo, hi);
        }
    }

    v8f o[4];
    float m8[8], l8[8];
    for (int i = 0; i < 4; ++i) o[i] = zero8();
    for (int r = 0; r < 8; ++r) { m8[r] = -1e30f; l8[r] = 0.0f; }

    const __bf16* Kg = Kb  + (size_t)(b * NKV + kvh) * SEQ * HD;
    const __bf16* Vg = Vbt + (size_t)(b * NKV + kvh) * HD * SEQ;
    lbf16* Kl3 = (lbf16*)&Kl[0];
    lbf16* Vt3 = (lbf16*)&Vt[0];

    const int ntiles = (qb + 1) * 2;
    for (int kt = 0; kt < ntiles; ++kt) {
        const int key0 = kt * 64;
        __syncthreads();                 // previous tile consumers done
        // K: 64 rows x 64 bf16; V: 64 d-rows x 64 keys -> 2+2 chunks/thread
        for (int i = 0; i < 2; ++i) {
            int c = tid + i * 256;       // 0..511
            int r = c >> 3;              // 0..63
            int j = c & 7;               // 0..7 (8 bf16 = 16B)
            cp16(Kl3 + r * 72 + j * 8, Kg + (size_t)(key0 + r) * HD + j * 8);
            cp16(Vt3 + r * 72 + j * 8, Vg + (size_t)r * SEQ + key0 + j * 8);
        }
        wait_async0();
        __syncthreads();                 // tile fully staged

        if (key0 <= qrow0 + 15) {        // tile not fully masked for this wave
            v8f s[4];
            for (int f4 = 0; f4 < 4; ++f4) {
                s[f4] = zero8();
                for (int dc = 0; dc < 2; ++dc) {
                    v16bf bk = lds_frag(&Kl[(f4 * 16 + l15) * 72 + dc * 32], lane);
                    s[f4] = __builtin_amdgcn_wmma_f32_16x16x32_bf16(
                        false, aq[dc], false, bk, (short)0, s[f4], false, false);
                }
            }
            // Mask only when the tile can cross the diagonal for this wave
            // (wave-uniform: skips the cmp/cndmask chains on interior tiles).
            if (key0 + 63 > qrow0) {
                for (int f4 = 0; f4 < 4; ++f4) {
                    int colg = key0 + f4 * 16 + l15;
                    for (int r = 0; r < 8; ++r)
                        if (colg > qrow0 + half * 8 + r) s[f4][r] = -1e30f;
                }
            }
            for (int r = 0; r < 8; ++r) {
                float v = fmaxf(fmaxf(s[0][r], s[1][r]), fmaxf(s[2][r], s[3][r]));
                v = fmaxf(v, __shfl_xor(v, 1, 32));
                v = fmaxf(v, __shfl_xor(v, 2, 32));
                v = fmaxf(v, __shfl_xor(v, 4, 32));
                v = fmaxf(v, __shfl_xor(v, 8, 32));
                float mn   = fmaxf(m8[r], v);
                float corr = __expf(m8[r] - mn);
                float psum = 0.0f;
                for (int f4 = 0; f4 < 4; ++f4) {
                    float p = __expf(s[f4][r] - mn);
                    s[f4][r] = p;
                    psum += p;
                }
                psum += __shfl_xor(psum, 1, 32);
                psum += __shfl_xor(psum, 2, 32);
                psum += __shfl_xor(psum, 4, 32);
                psum += __shfl_xor(psum, 8, 32);
                l8[r] = l8[r] * corr + psum;
                m8[r] = mn;
                for (int ni = 0; ni < 4; ++ni) o[ni][r] *= corr;
            }
            __bf16* Pw = &Pl[w][0];
            for (int f4 = 0; f4 < 4; ++f4)
                for (int r = 0; r < 8; ++r)
                    Pw[(half * 8 + r) * 72 + f4 * 16 + l15] = (__bf16)s[f4][r];
            for (int kc = 0; kc < 2; ++kc) {
                v16bf ap = lds_frag(&Pw[l15 * 72 + kc * 32], lane);
                for (int ni = 0; ni < 4; ++ni) {
                    v16bf bv = lds_frag(&Vt[(ni * 16 + l15) * 72 + kc * 32], lane);
                    o[ni] = __builtin_amdgcn_wmma_f32_16x16x32_bf16(
                        false, ap, false, bv, (short)0, o[ni], false, false);
                }
            }
        }
    }

    for (int r = 0; r < 8; ++r) {
        float inv = 1.0f / l8[r];
        int row = qrow0 + half * 8 + r;
        __bf16* op = Ob + (((size_t)b * SEQ + row) * NH + h) * HD;
        for (int ni = 0; ni < 4; ++ni)
            op[ni * 16 + l15] = (__bf16)(o[ni][r] * inv);
    }
}

// ---------------------------------------------------------------------------
// Launcher
// ---------------------------------------------------------------------------
extern "C" void kernel_launch(void* const* d_in, const int* in_sizes, int n_in,
                              void* d_out, int out_size, void* d_ws, size_t ws_size,
                              hipStream_t stream)
{
    (void)in_sizes; (void)n_in; (void)out_size; (void)ws_size;

    const float* x  = (const float*)d_in[0];
    const float* cs = (const float*)d_in[1];
    const float* sn = (const float*)d_in[2];
    const float* wq = (const float*)d_in[3];
    const float* wk = (const float*)d_in[4];
    const float* wv = (const float*)d_in[5];
    const float* wo = (const float*)d_in[6];

    float* out   = (float*)d_out;
    float* new_k = out   + (size_t)BATCH * SEQ * DIM;
    float* new_v = new_k + (size_t)BATCH * NKV * SEQ * HD;

    const int M = BATCH * SEQ;               // 4096
    const int KVD = NKV * HD;                // 512

    // workspace layout
    char* base = (char*)d_ws;
    __bf16* xb   = (__bf16*)base;                    base += (size_t)M * DIM * 2;
    __bf16* wqt  = (__bf16*)base;                    base += (size_t)DIM * DIM * 2;
    __bf16* wkt  = (__bf16*)base;                    base += (size_t)KVD * DIM * 2;
    __bf16* wvt  = (__bf16*)base;                    base += (size_t)KVD * DIM * 2;
    __bf16* wot  = (__bf16*)base;                    base += (size_t)DIM * DIM * 2;
    __bf16* qpb  = (__bf16*)base;                    base += (size_t)M * DIM * 2;   // q proj bf16
    float*  kpf  = (float*)base;                     base += (size_t)M * KVD * 4;   // k proj fp32
    float*  vpf  = (float*)base;                     base += (size_t)M * KVD * 4;   // v proj fp32
    __bf16* qTb  = (__bf16*)base;                    base += (size_t)BATCH * NH  * SEQ * HD * 2;
    __bf16* kbb  = (__bf16*)base;                    base += (size_t)BATCH * NKV * SEQ * HD * 2;
    __bf16* vbt  = (__bf16*)base;                    base += (size_t)BATCH * NKV * HD * SEQ * 2;
    __bf16* aob  = qpb;  // attention output reuses q-projection space

    // ---- pre-passes: convert / transpose to bf16 ----
    cvt_bf16<<<(M * DIM / 4) / 256, 256, 0, stream>>>(x, xb, M * DIM / 4);
    transpose_cvt_w<<<dim3(DIM / 32, DIM / 32), dim3(32, 8), 0, stream>>>(wq, wqt, DIM, DIM);
    transpose_cvt_w<<<dim3(KVD / 32, DIM / 32), dim3(32, 8), 0, stream>>>(wk, wkt, DIM, KVD);
    transpose_cvt_w<<<dim3(KVD / 32, DIM / 32), dim3(32, 8), 0, stream>>>(wv, wvt, DIM, KVD);
    transpose_cvt_w<<<dim3(DIM / 32, DIM / 32), dim3(32, 8), 0, stream>>>(wo, wot, DIM, DIM);

    // ---- QKV projections (q -> bf16, k/v -> fp32 for exact new_k/new_v path)
    gemm_bf16_t<__bf16><<<dim3(DIM / 128, M / 128), 256, 0, stream>>>(xb, wqt, qpb, DIM, DIM);
    gemm_bf16_t<float> <<<dim3(KVD / 128, M / 128), 256, 0, stream>>>(xb, wkt, kpf, KVD, DIM);
    gemm_bf16_t<float> <<<dim3(KVD / 128, M / 128), 256, 0, stream>>>(xb, wvt, vpf, KVD, DIM);

    // ---- RoPE / layout transforms ----
    rope_q<<<(BATCH * SEQ * NH * 32) / 256, 256, 0, stream>>>(qpb, qTb, cs, sn, BATCH * SEQ * NH * 32);
    rope_k<<<(BATCH * SEQ * NKV * 32) / 256, 256, 0, stream>>>(kpf, new_k, kbb, cs, sn, BATCH * SEQ * NKV * 32);
    copy_v<<<(BATCH * SEQ * NKV * 64) / 256, 256, 0, stream>>>(vpf, new_v, vbt, BATCH * SEQ * NKV * 64);

    // ---- flash attention ----
    attn_fwd<<<dim3(SEQ / 128, BATCH * NH), 256, 0, stream>>>(qTb, kbb, vbt, aob);

    // ---- output projection ----
    gemm_bf16_t<float><<<dim3(DIM / 128, M / 128), 256, 0, stream>>>(aob, wot, out, DIM, DIM);
}
